// QLoRALinear_40604620816621
// MI455X (gfx1250) — matrix operand order repaired and checked by reading, not compiled, not run on hardware
//
#include <hip/hip_runtime.h>
#include <hip/hip_bf16.h>

// QLoRA fused dequant-GEMM for MI455X (gfx1250, wave32, WMMA bf16).
// out = X @ (q*scale)^T + bias + 2.0 * (X @ A^T) @ B^T
//
// Fast path (needs ~96MB workspace):
//   pass 1: X  (f32)   -> bf16   (bandwidth-bound, 128MB read / 64MB write)
//   pass 2: qW (int32) -> bf16 dequant (64MB read / 32MB write)
//   pass 3: lora_A     -> bf16   (tiny)
//   pass 4: WMMA GEMM, tiles staged with global_load_async_to_lds_b128,
//           double-buffered LDS, zero conversion VALU in the hot loop.
// Fallback (small ws): single fused kernel with in-loop dequant.

typedef __attribute__((ext_vector_type(4)))  __bf16 v4bf;
typedef __attribute__((ext_vector_type(8)))  __bf16 v8bf;
typedef __attribute__((ext_vector_type(16))) __bf16 v16bf;
typedef __attribute__((ext_vector_type(8)))  float  v8f;

#define IN_F    4096
#define OUT_F   4096
#define TILE_M  128
#define TILE_N  128
#define TILE_K  32
#define LDS_K   40      // padded bf16 k-stride: 80B rows, keeps 16B-aligned sub-chunks
#define RANK    16
#define LORA_SCALING 2.0f

// ---------------- CDNA5 helpers ----------------

// Build a 16-elem bf16 fragment from two contiguous 16B LDS runs (ds_load_b128 x2).
__device__ __forceinline__ v16bf frag2(const __bf16* p0, const __bf16* p1) {
  v8bf lo = *(const v8bf*)p0;
  v8bf hi = *(const v8bf*)p1;
  return __builtin_shufflevector(lo, hi, 0,1,2,3,4,5,6,7,8,9,10,11,12,13,14,15);
}

__device__ __forceinline__ v8f wmma_bf16(v16bf a, v16bf b, v8f c) {
  // 8 args: (neg_a, A, neg_b, B, c_mod, C, reuse_a, reuse_b)
  return __builtin_amdgcn_wmma_f32_16x16x32_bf16(false, a, false, b, (short)0, c, false, false);
}

// Async copy 16B/lane global -> LDS, tracked with ASYNCcnt (no VGPR data path).
__device__ __forceinline__ void async_ld_b128(__bf16* lds, const __bf16* g) {
  uint32_t l = (uint32_t)(uintptr_t)lds;       // low 32 bits = LDS byte offset
  uint64_t a = (uint64_t)(uintptr_t)g;
  asm volatile("global_load_async_to_lds_b128 %0, %1, off"
               :: "v"(l), "v"(a) : "memory");
}

__device__ __forceinline__ void wait_async0() {
  asm volatile("s_wait_asynccnt 0x0" ::: "memory");
}

// ---------------- pre-pass kernels ----------------

__global__ __launch_bounds__(256)
void cvt_f32_to_bf16(const float* __restrict__ src, __bf16* __restrict__ dst) {
  size_t i = (size_t)blockIdx.x * blockDim.x + threadIdx.x;   // one float4 each
  float4 v = ((const float4*)src)[i];
  v4bf o = { (__bf16)v.x, (__bf16)v.y, (__bf16)v.z, (__bf16)v.w };
  ((v4bf*)dst)[i] = o;
}

__global__ __launch_bounds__(256)
void dequant_w_bf16(const int* __restrict__ qw, const float* __restrict__ absmax,
                    __bf16* __restrict__ dst) {
  size_t i = (size_t)blockIdx.x * blockDim.x + threadIdx.x;   // one int4 each
  int4 q = ((const int4*)qw)[i];
  size_t e = i * 4;
  int o = (int)(e >> 12);          // /4096
  int k = (int)(e & 4095);
  float s = absmax[o * (IN_F / 64) + (k >> 6)];
  v4bf d = { (__bf16)((float)q.x * s), (__bf16)((float)q.y * s),
             (__bf16)((float)q.z * s), (__bf16)((float)q.w * s) };
  ((v4bf*)dst)[i] = d;
}

// ---------------- fast-path GEMM: async-staged, double-buffered ----------------

__device__ __forceinline__ void stage_chunk_async(
    const __bf16* __restrict__ gX, const __bf16* __restrict__ gW,
    const __bf16* __restrict__ gA,
    __bf16* sX, __bf16* sW, __bf16* sA,
    int tid, int wave, int mbase, int nbase, int kb) {
#pragma unroll
  for (int i = 0; i < 2; ++i) {
    int idx = tid + i * 256;          // 0..511 chunk-units of 16B
    int row = idx >> 2;               // 0..127
    int c8  = (idx & 3) << 3;         // 0,8,16,24 (bf16 elems)
    async_ld_b128(sX + row * LDS_K + c8, gX + (size_t)(mbase + row) * IN_F + kb + c8);
    async_ld_b128(sW + row * LDS_K + c8, gW + (size_t)(nbase + row) * IN_F + kb + c8);
  }
  if (wave < 2) {                      // wave-uniform: waves 0,1 move lora_A (1KB)
    int row = tid >> 2;                // 0..15
    int c8  = (tid & 3) << 3;
    async_ld_b128(sA + row * LDS_K + c8, gA + (size_t)row * IN_F + kb + c8);
  }
}

__global__ __launch_bounds__(256)
void qlora_wmma_async(const __bf16* __restrict__ xb,
                      const __bf16* __restrict__ wb,
                      const __bf16* __restrict__ ab,
                      const float*  __restrict__ lB,
                      const float*  __restrict__ bias,
                      float* __restrict__ out) {
  __shared__ __align__(16) __bf16 Xs[2][TILE_M * LDS_K];
  __shared__ __align__(16) __bf16 Ws[2][TILE_N * LDS_K];
  __shared__ __align__(16) __bf16 As[2][RANK * LDS_K];
  __shared__ float aL[TILE_M * RANK];

  const int tid  = threadIdx.x;
  const int lane = tid & 31;
  const int wave = tid >> 5;
  const int wm   = wave >> 2;
  const int wn   = wave & 3;
  const int half = lane >> 4;
  const int l16  = lane & 15;

  const int mbase = blockIdx.y * TILE_M;
  const int nbase = blockIdx.x * TILE_N;

  v8f acc[4][2];
  v8f lacc[4];
  const v8f vzero = {0.f, 0.f, 0.f, 0.f, 0.f, 0.f, 0.f, 0.f};
#pragma unroll
  for (int mi = 0; mi < 4; ++mi) {
    lacc[mi] = vzero;
#pragma unroll
    for (int ni = 0; ni < 2; ++ni) acc[mi][ni] = vzero;
  }

  // prologue: fill buffer 0
  stage_chunk_async(xb, wb, ab, Xs[0], Ws[0], As[0], tid, wave, mbase, nbase, 0);
  wait_async0();
  __syncthreads();

  for (int kb = 0; kb < IN_F; kb += TILE_K) {
    const int buf = (kb >> 5) & 1;

    // kick async copy of the next chunk into the other buffer
    if (kb + TILE_K < IN_F)
      stage_chunk_async(xb, wb, ab, Xs[buf ^ 1], Ws[buf ^ 1], As[buf ^ 1],
                        tid, wave, mbase, nbase, kb + TILE_K);

    // A fragments: per-lane row l16, contiguous K run at half*16
    v16bf af[4];
#pragma unroll
    for (int mi = 0; mi < 4; ++mi) {
      const __bf16* base = &Xs[buf][(wm * 64 + mi * 16 + l16) * LDS_K];
      af[mi] = frag2(base + half * 16, base + half * 16 + 8);
    }
    // B fragments: per-lane column l16, contiguous K run at half*16
    v16bf bfr[2];
#pragma unroll
    for (int ni = 0; ni < 2; ++ni) {
      const __bf16* base = &Ws[buf][(wn * 32 + ni * 16 + l16) * LDS_K + half * 16];
      bfr[ni] = frag2(base, base + 8);
    }
#pragma unroll
    for (int mi = 0; mi < 4; ++mi)
#pragma unroll
      for (int ni = 0; ni < 2; ++ni)
        acc[mi][ni] = wmma_bf16(af[mi], bfr[ni], acc[mi][ni]);

    // LoRA rank-16 accumulation (wave-uniform branch; EXEC stays all-ones)
    if (wn == 0) {
      const __bf16* base = &As[buf][l16 * LDS_K + half * 16];
      v16bf lbf = frag2(base, base + 8);
#pragma unroll
      for (int mi = 0; mi < 4; ++mi)
        lacc[mi] = wmma_bf16(af[mi], lbf, lacc[mi]);
    }

    wait_async0();       // next buffer fully in LDS
    __syncthreads();     // all waves done reading this buffer / see async writes
  }

  // exchange X@A^T through LDS
  if (wn == 0) {
#pragma unroll
    for (int mi = 0; mi < 4; ++mi)
#pragma unroll
      for (int v = 0; v < 8; ++v)
        aL[(wm * 64 + mi * 16 + v + half * 8) * RANK + l16] = lacc[mi][v];
  }
  __syncthreads();

  // epilogue: out = acc + bias + 2.0 * aL @ lora_B^T
#pragma unroll
  for (int mi = 0; mi < 4; ++mi) {
#pragma unroll
    for (int ni = 0; ni < 2; ++ni) {
      int gcol = nbase + wn * 32 + ni * 16 + l16;
      float bv = bias[gcol];
      float lb[16];
      {
        const float4* p = (const float4*)(lB + (size_t)gcol * RANK);
        float4 b0 = p[0], b1 = p[1], b2 = p[2], b3 = p[3];
        lb[0]=b0.x; lb[1]=b0.y; lb[2]=b0.z; lb[3]=b0.w;
        lb[4]=b1.x; lb[5]=b1.y; lb[6]=b1.z; lb[7]=b1.w;
        lb[8]=b2.x; lb[9]=b2.y; lb[10]=b2.z; lb[11]=b2.w;
        lb[12]=b3.x; lb[13]=b3.y; lb[14]=b3.z; lb[15]=b3.w;
      }
#pragma unroll
      for (int v = 0; v < 8; ++v) {
        int lrow = wm * 64 + mi * 16 + v + half * 8;
        float s = 0.f;
#pragma unroll
        for (int r = 0; r < RANK; ++r) s += aL[lrow * RANK + r] * lb[r];
        int grow = mbase + lrow;
        out[(size_t)grow * OUT_F + gcol] = acc[mi][ni][v] + bv + LORA_SCALING * s;
      }
    }
  }
}

// ---------------- fallback: fused in-loop dequant (round-1 kernel) ----------------

__global__ __launch_bounds__(256)
void qlora_wmma_fused(const float* __restrict__ x,
                      const int*   __restrict__ qw,
                      const float* __restrict__ absmax,
                      const float* __restrict__ lA,
                      const float* __restrict__ lB,
                      const float* __restrict__ bias,
                      float* __restrict__ out) {
  __shared__ __align__(16) __bf16 Xs[TILE_M * LDS_K];
  __shared__ __align__(16) __bf16 Ws[TILE_N * LDS_K];
  __shared__ __align__(16) __bf16 As[RANK * LDS_K];
  __shared__ float aL[TILE_M * RANK];

  const int tid  = threadIdx.x;
  const int lane = tid & 31;
  const int wave = tid >> 5;
  const int wm   = wave >> 2;
  const int wn   = wave & 3;
  const int half = lane >> 4;
  const int l16  = lane & 15;

  const int mbase = blockIdx.y * TILE_M;
  const int nbase = blockIdx.x * TILE_N;

  v8f acc[4][2];
  v8f lacc[4];
  const v8f vzero = {0.f, 0.f, 0.f, 0.f, 0.f, 0.f, 0.f, 0.f};
#pragma unroll
  for (int mi = 0; mi < 4; ++mi) {
    lacc[mi] = vzero;
#pragma unroll
    for (int ni = 0; ni < 2; ++ni) acc[mi][ni] = vzero;
  }

  for (int kb = 0; kb < IN_F; kb += TILE_K) {
    const int kblk = kb >> 6;
#pragma unroll
    for (int i = 0; i < 4; ++i) {
      int idx = tid + i * 256;
      int row = idx >> 3;
      int c4  = (idx & 7) << 2;
      float4 v = *(const float4*)(x + (size_t)(mbase + row) * IN_F + kb + c4);
      __bf16* d = &Xs[row * LDS_K + c4];
      d[0] = (__bf16)v.x; d[1] = (__bf16)v.y; d[2] = (__bf16)v.z; d[3] = (__bf16)v.w;
    }
#pragma unroll
    for (int i = 0; i < 4; ++i) {
      int idx = tid + i * 256;
      int n   = idx >> 3;
      int c4  = (idx & 7) << 2;
      int o   = nbase + n;
      int4 q  = *(const int4*)(qw + (size_t)o * IN_F + kb + c4);
      float s = absmax[o * (IN_F / 64) + kblk];
      __bf16* d = &Ws[n * LDS_K + c4];
      d[0] = (__bf16)((float)q.x * s);
      d[1] = (__bf16)((float)q.y * s);
      d[2] = (__bf16)((float)q.z * s);
      d[3] = (__bf16)((float)q.w * s);
    }
    if (tid < 128) {
      int row = tid >> 3;
      int c4  = (tid & 7) << 2;
      float4 v = *(const float4*)(lA + (size_t)row * IN_F + kb + c4);
      __bf16* d = &As[row * LDS_K + c4];
      d[0] = (__bf16)v.x; d[1] = (__bf16)v.y; d[2] = (__bf16)v.z; d[3] = (__bf16)v.w;
    }
    if (kb + TILE_K < IN_F) {
      int row = tid >> 1;
      __builtin_prefetch(x  + (size_t)(mbase + row) * IN_F + kb + TILE_K, 0, 3);
      __builtin_prefetch(qw + (size_t)(nbase + row) * IN_F + kb + TILE_K, 0, 3);
    }
    __syncthreads();

    v16bf af[4];
#pragma unroll
    for (int mi = 0; mi < 4; ++mi) {
      const __bf16* base = &Xs[(wm * 64 + mi * 16 + l16) * LDS_K];
      af[mi] = frag2(base + half * 16, base + half * 16 + 8);
    }
    v16bf bfr[2];
#pragma unroll
    for (int ni = 0; ni < 2; ++ni) {
      const __bf16* base = &Ws[(wn * 32 + ni * 16 + l16) * LDS_K + half * 16];
      bfr[ni] = frag2(base, base + 8);
    }
#pragma unroll
    for (int mi = 0; mi < 4; ++mi)
#pragma unroll
      for (int ni = 0; ni < 2; ++ni)
        acc[mi][ni] = wmma_bf16(af[mi], bfr[ni], acc[mi][ni]);

    if (wn == 0) {
      const __bf16* base = &As[l16 * LDS_K + half * 16];
      v16bf lbf = frag2(base, base + 8);
#pragma unroll
      for (int mi = 0; mi < 4; ++mi)
        lacc[mi] = wmma_bf16(af[mi], lbf, lacc[mi]);
    }
    __syncthreads();
  }

  if (wn == 0) {
#pragma unroll
    for (int mi = 0; mi < 4; ++mi)
#pragma unroll
      for (int v = 0; v < 8; ++v)
        aL[(wm * 64 + mi * 16 + v + half * 8) * RANK + l16] = lacc[mi][v];
  }
  __syncthreads();

#pragma unroll
  for (int mi = 0; mi < 4; ++mi) {
#pragma unroll
    for (int ni = 0; ni < 2; ++ni) {
      int gcol = nbase + wn * 32 + ni * 16 + l16;
      float bv = bias[gcol];
      float lb[16];
      {
        const float4* p = (const float4*)(lB + (size_t)gcol * RANK);
        float4 b0 = p[0], b1 = p[1], b2 = p[2], b3 = p[3];
        lb[0]=b0.x; lb[1]=b0.y; lb[2]=b0.z; lb[3]=b0.w;
        lb[4]=b1.x; lb[5]=b1.y; lb[6]=b1.z; lb[7]=b1.w;
        lb[8]=b2.x; lb[9]=b2.y; lb[10]=b2.z; lb[11]=b2.w;
        lb[12]=b3.x; lb[13]=b3.y; lb[14]=b3.z; lb[15]=b3.w;
      }
#pragma unroll
      for (int v = 0; v < 8; ++v) {
        int lrow = wm * 64 + mi * 16 + v + half * 8;
        float s = 0.f;
#pragma unroll
        for (int r = 0; r < RANK; ++r) s += aL[lrow * RANK + r] * lb[r];
        int grow = mbase + lrow;
        out[(size_t)grow * OUT_F + gcol] = acc[mi][ni][v] + bv + LORA_SCALING * s;
      }
    }
  }
}

// ---------------- launcher ----------------

extern "C" void kernel_launch(void* const* d_in, const int* in_sizes, int n_in,
                              void* d_out, int out_size, void* d_ws, size_t ws_size,
                              hipStream_t stream) {
  const float* x    = (const float*)d_in[0];
  const int*   qw   = (const int*)d_in[1];
  const float* absm = (const float*)d_in[2];
  const float* lA   = (const float*)d_in[3];
  const float* lB   = (const float*)d_in[4];
  const float* bias = (const float*)d_in[5];
  float* out = (float*)d_out;

  const int M = in_sizes[0] / IN_F;  // 8192 tokens
  dim3 grid(OUT_F / TILE_N, M / TILE_M);

  const size_t xb_bytes = (size_t)M * IN_F * sizeof(__bf16);       // 64MB
  const size_t wb_bytes = (size_t)OUT_F * IN_F * sizeof(__bf16);   // 32MB
  const size_t ab_bytes = (size_t)RANK * IN_F * sizeof(__bf16);    // 128KB

  if (ws_size >= xb_bytes + wb_bytes + ab_bytes) {
    __bf16* xb = (__bf16*)d_ws;
    __bf16* wb = (__bf16*)((char*)d_ws + xb_bytes);
    __bf16* ab = (__bf16*)((char*)d_ws + xb_bytes + wb_bytes);

    const size_t nx4 = (size_t)M * IN_F / 4;
    cvt_f32_to_bf16<<<(unsigned)(nx4 / 256), 256, 0, stream>>>(x, xb);
    const size_t nw4 = (size_t)OUT_F * IN_F / 4;
    dequant_w_bf16<<<(unsigned)(nw4 / 256), 256, 0, stream>>>(qw, absm, wb);
    const size_t na4 = (size_t)RANK * IN_F / 4;
    cvt_f32_to_bf16<<<(unsigned)(na4 / 256), 256, 0, stream>>>(lA, ab);

    qlora_wmma_async<<<grid, dim3(256), 0, stream>>>(xb, wb, ab, lB, bias, out);
  } else {
    qlora_wmma_fused<<<grid, dim3(256), 0, stream>>>(x, qw, absm, lA, lB, bias, out);
  }
}
